// BertAdapter_35880156791633
// MI455X (gfx1250) — matrix operand order repaired and checked by reading, not compiled, not run on hardware
//
#include <hip/hip_runtime.h>
#include <math.h>

#define HIDDEN  1024
#define ADAPTER 256
#define MT      64        // tokens per block
#define MTILES  4         // MT/16
#define LDA     1032      // halves per LN row in LDS (516 dwords ≡ 4 mod 64 -> conflict-free frags)
#define LDH     264       // halves per GELU row in LDS (132 dwords ≡ 4 mod 64)

typedef __attribute__((ext_vector_type(16))) _Float16 v16h;
typedef __attribute__((ext_vector_type(8)))  float    v8f;

// ---------------------------------------------------------------------------
// Pre-pack both weight matrices fp32 -> f16 in WMMA B-fragment order:
//   packed[((ktile*NT + ntile)*32 + lane)*16 + e] = W[ktile*32 + (lane>>4)*8 + (e&7) + (e>>3)*16][ntile*16 + (lane&15)]
// so each lane's 16-half B operand is two contiguous 16B loads.
// w_down: 32 ktiles x 16 ntiles; w_up: 8 ktiles x 64 ntiles. 262144 elems each.
// ---------------------------------------------------------------------------
__global__ void pack_weights_kernel(const float* __restrict__ wd,
                                    const float* __restrict__ wu,
                                    _Float16* __restrict__ pd,
                                    _Float16* __restrict__ pu) {
    int idx  = blockIdx.x * blockDim.x + threadIdx.x;   // 0 .. 2*262144-1
    int half = (idx >= HIDDEN * ADAPTER) ? 1 : 0;
    int i    = half ? idx - HIDDEN * ADAPTER : idx;
    int e    = i & 15;
    int lane = (i >> 4) & 31;
    int blk  = i >> 9;                  // ktile*NT + ntile
    int k_in = ((lane >> 4) << 3) + (e & 7) + ((e >> 3) << 4);  // k within the 32-wide K tile
    int n    = lane & 15;
    if (!half) {
        int kt = blk >> 4, nt = blk & 15;
        pd[i] = (_Float16)wd[(size_t)(kt * 32 + k_in) * ADAPTER + nt * 16 + n];
    } else {
        int kt = blk >> 6, nt = blk & 63;
        pu[i] = (_Float16)wu[(size_t)(kt * 32 + k_in) * HIDDEN + nt * 16 + n];
    }
}

__device__ __forceinline__ float gelu_new_f(float u) {
    return 0.5f * u * (1.0f + tanhf(0.7978845608028654f * (u + 0.044715f * u * u * u)));
}

// ---------------------------------------------------------------------------
// Fully fused: LN -> GEMM1(+bias,GELU) -> GEMM2(+bias) -> +residual
// One block = 64 tokens (4 M-tiles), 8 wave32s. LDS for the LN tile (132KB)
// is reused for the GELU intermediate after GEMM1's K loop completes.
// All hot-loop addresses are hoisted base pointers + compile-time immediates.
// ---------------------------------------------------------------------------
__global__ __launch_bounds__(256)
void adapter_fused_kernel(const float* __restrict__ x,
                          const float* __restrict__ lnw,
                          const float* __restrict__ lnb,
                          const _Float16* __restrict__ pd,
                          const float* __restrict__ b_down,
                          const _Float16* __restrict__ pu,
                          const float* __restrict__ b_up,
                          float* __restrict__ out) {
    __shared__ _Float16 smem[MT * LDA];        // 132,096 B
    _Float16* lds_a  = smem;                   // phase 1/2: LN'd activations [MT][LDA]
    _Float16* lds_h1 = smem;                   // phase 2/3: GELU output     [MT][LDH] (alias)

    const int tid  = threadIdx.x;
    const int lane = tid & 31;
    const int wave = tid >> 5;
    const size_t row0 = (size_t)blockIdx.x * MT;

    // ---------------- Phase 1: LayerNorm fp32 -> f16 into LDS -------------
    #pragma unroll
    for (int rr = 0; rr < MT / 8; ++rr) {
        const int row = wave * (MT / 8) + rr;
        const float4* xr = (const float4*)(x + (row0 + row) * HIDDEN);
        float4 v[8];
        float s = 0.f, s2 = 0.f;
        #pragma unroll
        for (int i = 0; i < 8; ++i) {
            v[i] = xr[i * 32 + lane];
            s  += v[i].x + v[i].y + v[i].z + v[i].w;
            s2 += v[i].x * v[i].x + v[i].y * v[i].y + v[i].z * v[i].z + v[i].w * v[i].w;
        }
        #pragma unroll
        for (int off = 16; off > 0; off >>= 1) {
            s  += __shfl_xor(s,  off, 32);
            s2 += __shfl_xor(s2, off, 32);
        }
        const float mean = s * (1.f / HIDDEN);
        const float var  = s2 * (1.f / HIDDEN) - mean * mean;
        const float rstd = rsqrtf(var + 1e-5f);
        #pragma unroll
        for (int i = 0; i < 8; ++i) {
            const int c = (i * 32 + lane) * 4;
            const float4 w4 = ((const float4*)lnw)[i * 32 + lane];
            const float4 b4 = ((const float4*)lnb)[i * 32 + lane];
            _Float16* dst = &lds_a[row * LDA + c];
            dst[0] = (_Float16)((v[i].x - mean) * rstd * w4.x + b4.x);
            dst[1] = (_Float16)((v[i].y - mean) * rstd * w4.y + b4.y);
            dst[2] = (_Float16)((v[i].z - mean) * rstd * w4.z + b4.z);
            dst[3] = (_Float16)((v[i].w - mean) * rstd * w4.w + b4.w);
        }
    }
    __syncthreads();

    // A-fragment addressing (ISA 16-bit A layout, wave32):
    //   lanes 0-15: M=lane, K groups {0..7,16..23}; lanes 16-31: M=lane-16, K {8..15,24..31}
    const int m_a  = lane & 15;
    const int kh_a = (lane >> 4) * 4;   // dword offset of this half-wave's K group
    const int m_c  = (lane >> 4) * 8;   // C/D: row base for this half-wave
    const int n_c  = lane & 15;         // C/D: column

    // ---------------- Phase 2: GEMM1 [64x1024]x[1024x256] + GELU ----------
    // Each wave: 2 N-tiles x 4 M-tiles. B fragments reused across M.
    // Hoisted bases: abase[mt] folds row + kh_a; weight base folds (wave,lane).
    v8f acc1[MTILES][2];
    #pragma unroll
    for (int mt = 0; mt < MTILES; ++mt) {
        acc1[mt][0] = (v8f){}; acc1[mt][1] = (v8f){};
    }
    const int n0 = wave * 2, n1 = n0 + 1;
    {
        const unsigned* abase[MTILES];
        #pragma unroll
        for (int mt = 0; mt < MTILES; ++mt)
            abase[mt] = (const unsigned*)&lds_a[(mt * 16 + m_a) * LDA] + kh_a;
        // pd block (kt, nt) for this lane lives at pdw + kt*16384 + (nt-n0)*1024 bytes
        const char* pdw = (const char*)pd + ((size_t)(n0 * 32 + lane) * 16) * sizeof(_Float16);

        #pragma unroll
        for (int kt = 0; kt < 32; ++kt) {
            union { v16h v; uint4 q[2]; } a[MTILES];
            #pragma unroll
            for (int mt = 0; mt < MTILES; ++mt) {
                a[mt].q[0] = *(const uint4*)(abase[mt] + kt * 16);
                a[mt].q[1] = *(const uint4*)(abase[mt] + kt * 16 + 8);
            }
            union { v16h v; uint4 q[2]; } b0, b1;
            b0.q[0] = *(const uint4*)(pdw + kt * 16384);
            b0.q[1] = *(const uint4*)(pdw + kt * 16384 + 16);
            b1.q[0] = *(const uint4*)(pdw + kt * 16384 + 1024);
            b1.q[1] = *(const uint4*)(pdw + kt * 16384 + 1040);
            #pragma unroll
            for (int mt = 0; mt < MTILES; ++mt) {
                acc1[mt][0] = __builtin_amdgcn_wmma_f32_16x16x32_f16(false, a[mt].v, false, b0.v,
                                                                     (short)0, acc1[mt][0], false, false);
                acc1[mt][1] = __builtin_amdgcn_wmma_f32_16x16x32_f16(false, a[mt].v, false, b1.v,
                                                                     (short)0, acc1[mt][1], false, false);
            }
        }
    }
    __syncthreads();   // all reads of lds_a complete before aliasing as lds_h1

    #pragma unroll
    for (int t = 0; t < 2; ++t) {
        const int col = (t ? n1 : n0) * 16 + n_c;
        const float bias = b_down[col];
        #pragma unroll
        for (int mt = 0; mt < MTILES; ++mt) {
            #pragma unroll
            for (int j = 0; j < 8; ++j) {
                lds_h1[(mt * 16 + m_c + j) * LDH + col] =
                    (_Float16)gelu_new_f(acc1[mt][t][j] + bias);
            }
        }
    }
    __syncthreads();

    // ---------------- Phase 3: GEMM2 [64x256]x[256x1024] + residual -------
    // nt-pair outer loop keeps 8 accumulators live; pu read exactly once per block.
    const unsigned* hbase[MTILES];
    #pragma unroll
    for (int mt = 0; mt < MTILES; ++mt)
        hbase[mt] = (const unsigned*)&lds_h1[(mt * 16 + m_a) * LDH] + kh_a;
    // pu block (kt, wave*8+q) for this lane: puw + kt*65536 + q*1024 bytes
    const char* puw = (const char*)pu + ((size_t)(wave * 8 * 32 + lane) * 16) * sizeof(_Float16);

    #pragma unroll
    for (int np = 0; np < 4; ++np) {
        const int qA = np * 2, qB = qA + 1;
        v8f acc2[MTILES][2];
        #pragma unroll
        for (int mt = 0; mt < MTILES; ++mt) {
            acc2[mt][0] = (v8f){}; acc2[mt][1] = (v8f){};
        }
        #pragma unroll
        for (int kt = 0; kt < 8; ++kt) {
            union { v16h v; uint4 q[2]; } a[MTILES];
            #pragma unroll
            for (int mt = 0; mt < MTILES; ++mt) {
                a[mt].q[0] = *(const uint4*)(hbase[mt] + kt * 16);
                a[mt].q[1] = *(const uint4*)(hbase[mt] + kt * 16 + 8);
            }
            union { v16h v; uint4 q[2]; } bA, bB;
            bA.q[0] = *(const uint4*)(puw + kt * 65536 + qA * 1024);
            bA.q[1] = *(const uint4*)(puw + kt * 65536 + qA * 1024 + 16);
            bB.q[0] = *(const uint4*)(puw + kt * 65536 + qB * 1024);
            bB.q[1] = *(const uint4*)(puw + kt * 65536 + qB * 1024 + 16);
            #pragma unroll
            for (int mt = 0; mt < MTILES; ++mt) {
                acc2[mt][0] = __builtin_amdgcn_wmma_f32_16x16x32_f16(false, a[mt].v, false, bA.v,
                                                                     (short)0, acc2[mt][0], false, false);
                acc2[mt][1] = __builtin_amdgcn_wmma_f32_16x16x32_f16(false, a[mt].v, false, bB.v,
                                                                     (short)0, acc2[mt][1], false, false);
            }
        }
        #pragma unroll
        for (int t = 0; t < 2; ++t) {
            const int col = (wave * 8 + (t ? qB : qA)) * 16 + n_c;
            const float bias = b_up[col];
            #pragma unroll
            for (int mt = 0; mt < MTILES; ++mt) {
                #pragma unroll
                for (int j = 0; j < 8; ++j) {
                    const size_t off = (row0 + mt * 16 + m_c + j) * HIDDEN + col;
                    out[off] = acc2[mt][t][j] + bias + x[off];  // residual re-read hits L2
                }
            }
        }
    }
}

extern "C" void kernel_launch(void* const* d_in, const int* in_sizes, int n_in,
                              void* d_out, int out_size, void* d_ws, size_t ws_size,
                              hipStream_t stream) {
    const float* x   = (const float*)d_in[0];
    const float* lnw = (const float*)d_in[1];
    const float* lnb = (const float*)d_in[2];
    const float* wd  = (const float*)d_in[3];
    const float* bd  = (const float*)d_in[4];
    const float* wu  = (const float*)d_in[5];
    const float* bu  = (const float*)d_in[6];
    float* out = (float*)d_out;

    _Float16* pd = (_Float16*)d_ws;
    _Float16* pu = pd + (size_t)HIDDEN * ADAPTER;

    // 2*262144 packed elements, 256 threads/block
    pack_weights_kernel<<<2048, 256, 0, stream>>>(wd, wu, pd, pu);

    const int tokens = in_sizes[0] / HIDDEN;      // 8*4096 = 32768
    adapter_fused_kernel<<<tokens / MT, 256, 0, stream>>>(x, lnw, lnb, pd, bd, pu, bu, out);
}
